// GRUD_Layer_85968065397120
// MI455X (gfx1250) — matrix operand order repaired and checked by reading, not compiled, not run on hardware
//
#include <hip/hip_runtime.h>
#include <hip/hip_bf16.h>

#define NINP    128
#define NHID    512
#define CDIM    768     // NINP + NHID + NINP
#define BATCH   512
#define TSTEPS  256
#define MT      16      // batch rows per workgroup (WMMA M)
#define NTHREADS 256    // 8 waves (wave32)
#define CP      776     // combined row stride (bf16): 388 dwords == 4 (mod 64)
#define DP      136     // past-tile row stride (bf16): 68 dwords == 4 (mod 64)
#define HP      520     // h row stride (f32)

typedef __bf16 bf16;
typedef __attribute__((ext_vector_type(16))) __bf16 v16bf;
typedef __attribute__((ext_vector_type(8)))  __bf16 v8bf;
typedef __attribute__((ext_vector_type(8)))  float  v8f;

__device__ __forceinline__ v16bf make16(v8bf lo, v8bf hi) {
  return __builtin_shufflevector(lo, hi, 0,1,2,3,4,5,6,7,8,9,10,11,12,13,14,15);
}

// A fragment (16x32 bf16), ISA 7.12.2 16-bit A layout:
// lanes 0-15 : K = k0+0..7 (elems 0-7), k0+16..23 (elems 8-15)
// lanes 16-31: K = k0+8..15,            k0+24..31
__device__ __forceinline__ v16bf load_fragA(const bf16* base, int stride, int k0, int lane) {
  const bf16* p = base + (lane & 15) * stride + k0 + ((lane >> 4) << 3);
  v8bf lo = *(const v8bf*)p;
  v8bf hi = *(const v8bf*)(p + 16);
  return make16(lo, hi);
}

// B fragment (32x16 bf16), B[k][n] = W[n0+n][k0+k], W row-major (stride in elems).
// lane = column n; lanes 0-15 hold K = k0+0..7, k0+16..23; lanes 16-31 K = k0+8..15, k0+24..31.
__device__ __forceinline__ v16bf load_fragB(const bf16* w, int stride, int k0, int lane) {
  const bf16* p = w + (lane & 15) * stride + k0 + ((lane >> 4) << 3);
  v8bf lo = *(const v8bf*)p;
  v8bf hi = *(const v8bf*)(p + 16);
  return make16(lo, hi);
}

__device__ __forceinline__ v8f wmma_bf16(v16bf a, v16bf b, v8f c) {
  return __builtin_amdgcn_wmma_f32_16x16x32_bf16(false, a, false, b, (short)0, c, false, false);
}

__device__ __forceinline__ float sigmoidf_(float x) { return 1.f / (1.f + __expf(-x)); }

__global__ void f32_to_bf16(const float* __restrict__ src, bf16* __restrict__ dst, int n) {
  int i = blockIdx.x * 256 + threadIdx.x;
  if (i < n) dst[i] = (bf16)src[i];
}

__global__ __launch_bounds__(NTHREADS)
void grud_persistent(const float* __restrict__ vals, const float* __restrict__ masks,
                     const float* __restrict__ past,
                     const float* __restrict__ bz, const float* __restrict__ br,
                     const float* __restrict__ bh, const float* __restrict__ bgx,
                     const float* __restrict__ bgh,
                     const bf16* __restrict__ Wz, const bf16* __restrict__ Wr,
                     const bf16* __restrict__ Wh, const bf16* __restrict__ Wgx,
                     const bf16* __restrict__ Wgh,
                     float* __restrict__ out) {
  __shared__ bf16  sComb[MT][CP];        // [xt | h/r*h | m] bf16
  __shared__ bf16  sD[MT][DP];           // past tile bf16 (GEMM A operand)
  __shared__ float sHf[MT][HP];          // h state f32
  __shared__ float sDx[MT][NINP + 4];    // delta_x f32
  __shared__ float sBz[NHID], sBr[NHID], sBh[NHID], sBgh[NHID], sBgx[NINP];

  const int tid  = threadIdx.x;
  const int wave = tid >> 5;
  const int lane = tid & 31;
  const int lhalf = lane >> 4;
  const int nl   = lane & 15;
  const int b0   = blockIdx.x * MT;

  for (int i = tid; i < NHID; i += NTHREADS) {
    sBz[i] = bz[i]; sBr[i] = br[i]; sBh[i] = bh[i]; sBgh[i] = bgh[i];
  }
  for (int i = tid; i < NINP; i += NTHREADS) sBgx[i] = bgx[i];
  for (int i = tid; i < MT * CP; i += NTHREADS) (&sComb[0][0])[i] = (bf16)0.f;
  for (int i = tid; i < MT * HP; i += NTHREADS) (&sHf[0][0])[i] = 0.f;

  // elementwise ownership: 16 threads per batch row, 8 contiguous features each
  const int erow = tid >> 4;
  const int ef0  = (tid & 15) << 3;
  float cm[8], cmv[8], xp[8];
#pragma unroll
  for (int j = 0; j < 8; ++j) { cm[j] = 0.f; cmv[j] = 0.f; xp[j] = 0.f; }

  const size_t ibase = ((size_t)(b0 + erow) * TSTEPS) * NINP + ef0;
  const float* vptr = vals  + ibase;
  const float* mptr = masks + ibase;
  const float* dptr = past  + ibase;

  __syncthreads();

  for (int t = 0; t < TSTEPS; ++t) {
    // ---- phase 1: load inputs for this timestep
    float vv[8], mm[8], dd[8];
    const float* vp = vptr + (size_t)t * NINP;
    const float* mp = mptr + (size_t)t * NINP;
    const float* dp = dptr + (size_t)t * NINP;
#pragma unroll
    for (int j = 0; j < 8; ++j) { vv[j] = vp[j]; mm[j] = mp[j]; dd[j] = dp[j]; }
    if (t + 1 < TSTEPS) {   // global_prefetch_b8 for next step's rows
      __builtin_prefetch(vp + NINP, 0, 0);
      __builtin_prefetch(mp + NINP, 0, 0);
      __builtin_prefetch(dp + NINP, 0, 0);
    }
#pragma unroll
    for (int j = 0; j < 8; ++j) {
      sD[erow][ef0 + j] = (bf16)dd[j];
      sComb[erow][NINP + NHID + ef0 + j] = (bf16)mm[j];
    }
    __syncthreads();  // (1) sD + mask region ready

    // ---- phase 2: delta_h = exp(-relu(d@Wgh^T+bgh)); h *= delta_h; delta_x
#pragma unroll
    for (int jt = 0; jt < 4; ++jt) {
      const int n0 = (wave * 4 + jt) * 16;
      v8f acc = {};
#pragma unroll
      for (int k = 0; k < NINP; k += 32) {
        v16bf a = load_fragA(&sD[0][0], DP, k, lane);
        v16bf b = load_fragB(Wgh + (size_t)n0 * NINP, NINP, k, lane);
        acc = wmma_bf16(a, b, acc);
      }
      const int n = n0 + nl;
      const float bias = sBgh[n];
#pragma unroll
      for (int i = 0; i < 8; ++i) {
        const int mr = lhalf * 8 + i;
        float dh = __expf(-fmaxf(acc[i] + bias, 0.f));
        float hn = dh * sHf[mr][n];
        sHf[mr][n] = hn;
        sComb[mr][NINP + n] = (bf16)hn;
      }
    }
    {
      const int n0 = wave * 16;   // 8 waves cover all 128 delta_x columns
      v8f acc = {};
#pragma unroll
      for (int k = 0; k < NINP; k += 32) {
        v16bf a = load_fragA(&sD[0][0], DP, k, lane);
        v16bf b = load_fragB(Wgx + (size_t)n0 * NINP, NINP, k, lane);
        acc = wmma_bf16(a, b, acc);
      }
      const int n = n0 + nl;
      const float bias = sBgx[n];
#pragma unroll
      for (int i = 0; i < 8; ++i)
        sDx[lhalf * 8 + i][n] = __expf(-fmaxf(acc[i] + bias, 0.f));
    }
    __syncthreads();  // (2) delta_x, scaled h ready

    // ---- phase 3: running-mean scan, x_prime scan, x_t
#pragma unroll
    for (int j = 0; j < 8; ++j) {
      cm[j]  += mm[j];
      cmv[j] += mm[j] * vv[j];
      float xm = (cm[j] > 0.f) ? (cmv[j] / cm[j]) : 0.f;
      xp[j] = mm[j] * vv[j] + (1.f - mm[j]) * xp[j];
      float dx = sDx[erow][ef0 + j];
      float xt = mm[j] * vv[j] + (1.f - mm[j]) * (dx * xp[j] + (1.f - dx) * xm);
      sComb[erow][ef0 + j] = (bf16)xt;
    }
    __syncthreads();  // (3) combined = [xt, h, m] ready

    // ---- phase 4: r = sigm(combined@Wr^T+br), z = sigm(combined@Wz^T+bz)
    v8f accr[4] = {}; v8f accz[4] = {};
    for (int k = 0; k < CDIM; k += 32) {
      v16bf a = load_fragA(&sComb[0][0], CP, k, lane);
#pragma unroll
      for (int jt = 0; jt < 4; ++jt) {
        const int n0 = wave * 64 + jt * 16;
        accr[jt] = wmma_bf16(a, load_fragB(Wr + (size_t)n0 * CDIM, CDIM, k, lane), accr[jt]);
        accz[jt] = wmma_bf16(a, load_fragB(Wz + (size_t)n0 * CDIM, CDIM, k, lane), accz[jt]);
      }
    }
#pragma unroll
    for (int jt = 0; jt < 4; ++jt) {
      const int n = wave * 64 + jt * 16 + nl;
      const float brv = sBr[n], bzv = sBz[n];
#pragma unroll
      for (int i = 0; i < 8; ++i) {
        accr[jt][i] = sigmoidf_(accr[jt][i] + brv);
        accz[jt][i] = sigmoidf_(accz[jt][i] + bzv);
      }
    }
    __syncthreads();  // (4) all waves finished reading combined h-region
#pragma unroll
    for (int jt = 0; jt < 4; ++jt) {
      const int n = wave * 64 + jt * 16 + nl;
#pragma unroll
      for (int i = 0; i < 8; ++i) {
        const int mr = lhalf * 8 + i;
        sComb[mr][NINP + n] = (bf16)(accr[jt][i] * sHf[mr][n]);  // r*h
      }
    }
    __syncthreads();  // (5) new_combined ready

    // ---- phase 5: h_tilde = tanh(new_combined@Wh^T+bh); h = (1-z)h + z*h~
    v8f acch[4] = {};
    for (int k = 0; k < CDIM; k += 32) {
      v16bf a = load_fragA(&sComb[0][0], CP, k, lane);
#pragma unroll
      for (int jt = 0; jt < 4; ++jt) {
        const int n0 = wave * 64 + jt * 16;
        acch[jt] = wmma_bf16(a, load_fragB(Wh + (size_t)n0 * CDIM, CDIM, k, lane), acch[jt]);
      }
    }
    float* orow = out + ((size_t)t * BATCH + b0) * NHID;
#pragma unroll
    for (int jt = 0; jt < 4; ++jt) {
      const int n = wave * 64 + jt * 16 + nl;
      const float bias = sBh[n];
#pragma unroll
      for (int i = 0; i < 8; ++i) {
        const int mr = lhalf * 8 + i;
        float ht = tanhf(acch[jt][i] + bias);
        float z  = accz[jt][i];
        float hn = (1.f - z) * sHf[mr][n] + z * ht;
        sHf[mr][n] = hn;                       // (mr,n) uniquely owned by this lane
        orow[(size_t)mr * NHID + n] = hn;
      }
    }
    __syncthreads();  // (6) protect combined from next-step phase-1 overwrite
  }
}

extern "C" void kernel_launch(void* const* d_in, const int* in_sizes, int n_in,
                              void* d_out, int out_size, void* d_ws, size_t ws_size,
                              hipStream_t stream) {
  const float* vals  = (const float*)d_in[0];
  const float* masks = (const float*)d_in[1];
  const float* past  = (const float*)d_in[2];
  const float* Wz    = (const float*)d_in[3];
  const float* bzp   = (const float*)d_in[4];
  const float* Wr    = (const float*)d_in[5];
  const float* brp   = (const float*)d_in[6];
  const float* Wh    = (const float*)d_in[7];
  const float* bhp   = (const float*)d_in[8];
  const float* Wgx   = (const float*)d_in[9];
  const float* bgxp  = (const float*)d_in[10];
  const float* Wgh   = (const float*)d_in[11];
  const float* bghp  = (const float*)d_in[12];

  // workspace: bf16 copies of the weights (reconverted every call; deterministic)
  bf16* wsb   = (bf16*)d_ws;
  bf16* Wz_b  = wsb;                       // 512*768
  bf16* Wr_b  = wsb + 393216;              // 512*768
  bf16* Wh_b  = wsb + 786432;              // 512*768
  bf16* Wgx_b = wsb + 1179648;             // 128*128
  bf16* Wgh_b = wsb + 1196032;             // 512*128
  f32_to_bf16<<<(393216 + 255) / 256, 256, 0, stream>>>(Wz,  Wz_b,  393216);
  f32_to_bf16<<<(393216 + 255) / 256, 256, 0, stream>>>(Wr,  Wr_b,  393216);
  f32_to_bf16<<<(393216 + 255) / 256, 256, 0, stream>>>(Wh,  Wh_b,  393216);
  f32_to_bf16<<<(16384  + 255) / 256, 256, 0, stream>>>(Wgx, Wgx_b, 16384);
  f32_to_bf16<<<(65536  + 255) / 256, 256, 0, stream>>>(Wgh, Wgh_b, 65536);

  grud_persistent<<<BATCH / MT, NTHREADS, 0, stream>>>(
      vals, masks, past, bzp, brp, bhp, bgxp, bghp,
      Wz_b, Wr_b, Wh_b, Wgx_b, Wgh_b, (float*)d_out);
}
